// LinformerAttention_wind2_42588895707687
// MI455X (gfx1250) — matrix operand order repaired and checked by reading, not compiled, not run on hardware
//
#include <hip/hip_runtime.h>
#include <hip/hip_bf16.h>
#include <math.h>

// ---------------- problem constants ----------------
constexpr int B_ = 8, N_ = 8192, DIM_ = 256;
constexpr int H_ = 8, D_ = 32, KR_ = 64, WS_ = 256;
constexpr int G_ = N_ / WS_;            // 32 windows
constexpr int INNER_ = H_ * D_;         // 256
constexpr int QKVC_ = 3 * INNER_;       // 768
constexpr float SCALE_ = 0.17677669529663687f;  // 32^-0.5

// per-tensor element counts in workspace (bf16 elements)
constexpr long long QSZ_ = (long long)B_ * H_ * G_ * WS_ * D_;  // 16,777,216

typedef __attribute__((ext_vector_type(16))) __bf16 v16bf;
typedef __attribute__((ext_vector_type(8)))  float  v8f;

// ---------------- WMMA fragment loaders (CDNA5 16x16x32 bf16 layouts) ------
// A matrix 16x32 (MxK), row-major source with leading dim lda.
// lane L: row = L&15, half = L>>4; elems 0..15 -> K = half*16 + i.
__device__ __forceinline__ v16bf load_a_f32(const float* A, int lda) {
  int lane = threadIdx.x & 31;
  const float* p = A + (lane & 15) * lda + (lane >> 4) * 16;
  v16bf a;
#pragma unroll
  for (int i = 0; i < 8; ++i) a[i] = (__bf16)p[i];
#pragma unroll
  for (int i = 0; i < 8; ++i) a[8 + i] = (__bf16)p[8 + i];
  return a;
}

__device__ __forceinline__ v16bf load_a_bf16(const __bf16* A, int lda) {
  int lane = threadIdx.x & 31;
  const __bf16* p = A + (lane & 15) * lda + (lane >> 4) * 16;
  v16bf a;
#pragma unroll
  for (int i = 0; i < 8; ++i) a[i] = p[i];
#pragma unroll
  for (int i = 0; i < 8; ++i) a[8 + i] = p[8 + i];
  return a;
}

// B matrix 32x16 (KxN), row-major source with leading dim ldb.
// lane L: N = L&15, K = 16*(L>>4) + i for element i.
__device__ __forceinline__ v16bf load_b_bf16(const __bf16* Bm, int ldb) {
  int lane = threadIdx.x & 31;
  const __bf16* p = Bm + (lane >> 4) * 16 * ldb + (lane & 15);
  v16bf b;
#pragma unroll
  for (int i = 0; i < 16; ++i) b[i] = p[i * ldb];
  return b;
}

// B = W^T where W is row-major [Ncols x Ktot] fp32; caller pre-offsets W to
// (first_col*ldw + k0); each lane reads 16 contiguous floats of one W row.
__device__ __forceinline__ v16bf load_bT_f32(const float* W, int ldw) {
  int lane = threadIdx.x & 31;
  const float* p = W + (lane & 15) * ldw + (lane >> 4) * 16;
  v16bf b;
#pragma unroll
  for (int i = 0; i < 16; ++i) b[i] = (__bf16)p[i];
  return b;
}

__device__ __forceinline__ v8f wmma_bf16(v16bf a, v16bf b, v8f c) {
  return __builtin_amdgcn_wmma_f32_16x16x32_bf16(false, a, false, b, (short)0,
                                                 c, false, false);
}

// ---------------- Kernel 1: QKV projection --------------------------------
// qkv = x[65536x256] @ Wqkv^T[256x768]; each wave computes a 16x64 strip
// (1 A-frag -> 4 B-frags -> 4 wmma per K-step); scatter to q/k/v bf16.
__global__ void __launch_bounds__(256)
qkv_proj_kernel(const float* __restrict__ x, const float* __restrict__ Wqkv,
                unsigned short* __restrict__ qraw, unsigned short* __restrict__ kraw,
                unsigned short* __restrict__ vraw) {
  __bf16* qo = reinterpret_cast<__bf16*>(qraw);
  __bf16* ko = reinterpret_cast<__bf16*>(kraw);
  __bf16* vo = reinterpret_cast<__bf16*>(vraw);

  int wave = threadIdx.x >> 5;
  int lane = threadIdx.x & 31;
  int t = blockIdx.x * 8 + wave;          // 4096*12 = 49152 strip jobs
  int strip  = t % (QKVC_ / 64);          // 12 col strips of 64
  int tile_m = t / (QKVC_ / 64);          // 4096 row tiles

  v8f acc[4] = {};
#pragma unroll
  for (int k0 = 0; k0 < DIM_; k0 += 32) {
    v16bf a = load_a_f32(x + (long long)tile_m * 16 * DIM_ + k0, DIM_);
#pragma unroll
    for (int j = 0; j < 4; ++j) {
      v16bf b = load_bT_f32(Wqkv + (long long)(strip * 64 + j * 16) * DIM_ + k0, DIM_);
      acc[j] = wmma_bf16(a, b, acc[j]);
    }
  }

  int half = lane >> 4;
  int n_local = lane & 15;
#pragma unroll
  for (int j = 0; j < 4; ++j) {
    // each 16-col subtile sits in one (head, q/k/v) chunk: 96 = 6*16
    int c_base = strip * 64 + j * 16;
    int h = c_base / 96;
    int which = (c_base % 96) / 32;
    __bf16* dst = (which == 0) ? qo : (which == 1) ? ko : vo;
    int d = (c_base % 32) + n_local;
#pragma unroll
    for (int r = 0; r < 8; ++r) {
      int R = tile_m * 16 + r + 8 * half;
      int bb = R >> 13;               // / N_
      int n  = R & (N_ - 1);
      int g  = n >> 8;                // / WS_
      int w  = n & (WS_ - 1);
      long long off = ((((long long)bb * H_ + h) * G_ + g) * WS_ + w) * D_ + d;
      dst[off] = (__bf16)acc[j][r];
    }
  }
}

// ---------------- Kernel 2: fused windowed Linformer attention ------------
// one block per (b,h,g); 8 waves; async-stage kw/vw into LDS, then
// kp/vp (WMMA) -> dots (WMMA) -> softmax -> out (WMMA), all in LDS.
constexpr int KPT_LD = 72;   // [32][64] bf16, padded
constexpr int VP_LD  = 40;   // [64][32] bf16, padded
constexpr int DT_LD  = 72;   // [256][64] bf16, padded
constexpr int SMEM_ELEMS = 256 * DT_LD;  // 18432 bf16 = 36.9KB (>= kw+vw 16384)

__global__ void __launch_bounds__(256)
attn_kernel(const unsigned short* __restrict__ qraw,
            const unsigned short* __restrict__ kraw,
            const unsigned short* __restrict__ vraw,
            const float* __restrict__ E, const float* __restrict__ F,
            unsigned short* __restrict__ oraw) {
  const __bf16* q = reinterpret_cast<const __bf16*>(qraw);
  const __bf16* k = reinterpret_cast<const __bf16*>(kraw);
  const __bf16* v = reinterpret_cast<const __bf16*>(vraw);
  __bf16* outw = reinterpret_cast<__bf16*>(oraw);

  // smem is time-shared: [phase 0-1] kwS(8192) ++ vwS(8192)  [phase 2+] dots
  __shared__ __bf16 smem[SMEM_ELEMS];
  __shared__ __bf16 kpT[32 * KPT_LD];
  __shared__ __bf16 vp[64 * VP_LD];
  __bf16* kwS  = smem;                  // [WS][32] bf16
  __bf16* vwS  = smem + WS_ * D_;
  __bf16* dots = smem;                  // [256][DT_LD] bf16

  int blk = blockIdx.x;                 // b*H*G + h*G + g
  int g = blk % G_;
  int h = (blk / G_) % H_;
  int bb = blk / (G_ * H_);

  int wave = threadIdx.x >> 5;
  int lane = threadIdx.x & 31;
  int half = lane >> 4, n_local = lane & 15;

  long long win = ((((long long)bb * H_ + h) * G_ + g) * WS_) * D_;
  const __bf16* kw = k + win;           // [WS][32]
  const __bf16* vw = v + win;
  const __bf16* qw = q + win;

  // ---- phase 0: async copy kw,vw (16KB each) global -> LDS --------------
  {
    int tid = threadIdx.x;
    const char* kg = (const char*)kw;
    const char* vg = (const char*)vw;
    char* ks = (char*)kwS;
    char* vs = (char*)vwS;
#pragma unroll
    for (int r = 0; r < 4; ++r) {
      int off = r * 4096 + tid * 16;
      unsigned kl = (unsigned)(unsigned long long)(ks + off);  // LDS byte offset
      unsigned long long ka = (unsigned long long)(kg + off);
      asm volatile("global_load_async_to_lds_b128 %0, %1, off"
                   :: "v"(kl), "v"(ka) : "memory");
      unsigned vl = (unsigned)(unsigned long long)(vs + off);
      unsigned long long va = (unsigned long long)(vg + off);
      asm volatile("global_load_async_to_lds_b128 %0, %1, off"
                   :: "v"(vl), "v"(va) : "memory");
    }
    asm volatile("s_wait_asynccnt 0x0" ::: "memory");
  }
  __syncthreads();

  // ---- phase 1: kp = Ew @ kw  (64x32), vp = Fw @ vw ----
  {
    int mt = wave >> 1, nt = wave & 1;  // 4 x 2 tiles, one per wave
    const float* Ewb = E + ((long long)h * KR_ + mt * 16) * N_ + g * WS_;
    const float* Fwb = F + ((long long)h * KR_ + mt * 16) * N_ + g * WS_;

    v8f ak = {}, av = {};
#pragma unroll
    for (int k0 = 0; k0 < WS_; k0 += 32) {
      v16bf aE = load_a_f32(Ewb + k0, N_);
      v16bf bK = load_b_bf16(kwS + k0 * D_ + nt * 16, D_);
      ak = wmma_bf16(aE, bK, ak);
      v16bf aF = load_a_f32(Fwb + k0, N_);
      v16bf bV = load_b_bf16(vwS + k0 * D_ + nt * 16, D_);
      av = wmma_bf16(aF, bV, av);
    }
#pragma unroll
    for (int r = 0; r < 8; ++r) {
      int m = mt * 16 + r + 8 * half;
      int d = nt * 16 + n_local;
      kpT[d * KPT_LD + m] = (__bf16)ak[r];   // transposed for dots B operand
      vp[m * VP_LD + d]   = (__bf16)av[r];
    }
  }
  __syncthreads();     // also retires all reads of kwS/vwS before dots reuse

  // ---- phase 2: dots = qw[256x32] @ kp^T[32x64], scaled ----
  for (int s = wave; s < 16; s += 8) {    // 16 row strips, 2 per wave
    v16bf a = load_a_bf16(qw + s * 16 * D_, D_);
#pragma unroll
    for (int nt = 0; nt < 4; ++nt) {
      v16bf b = load_b_bf16(kpT + nt * 16, KPT_LD);
      v8f acc = {};
      acc = wmma_bf16(a, b, acc);
#pragma unroll
      for (int r = 0; r < 8; ++r) {
        int row = s * 16 + r + 8 * half;
        int col = nt * 16 + n_local;
        dots[row * DT_LD + col] = (__bf16)(acc[r] * SCALE_);
      }
    }
  }
  __syncthreads();

  // ---- phase 3: softmax over m=64, one row per thread, in place ----
  {
    int row = threadIdx.x;
    float vals[64];
    float mx = -3.0e38f;
#pragma unroll
    for (int j = 0; j < 64; ++j) {
      float xv = (float)dots[row * DT_LD + j];
      vals[j] = xv;
      mx = fmaxf(mx, xv);
    }
    float s = 0.f;
#pragma unroll
    for (int j = 0; j < 64; ++j) {
      float e = __expf(vals[j] - mx);
      vals[j] = e;
      s += e;
    }
    float inv = 1.0f / s;
#pragma unroll
    for (int j = 0; j < 64; ++j)
      dots[row * DT_LD + j] = (__bf16)(vals[j] * inv);
  }
  __syncthreads();

  // ---- phase 4: out = attn[256x64] @ vp[64x32] -> attn_out[b][n][h*32+d] ----
  for (int s = wave; s < 16; s += 8) {    // 16 row strips, 2 per wave
    v8f acc[2] = {};
#pragma unroll
    for (int k0 = 0; k0 < 64; k0 += 32) {
      v16bf a = load_a_bf16(dots + s * 16 * DT_LD + k0, DT_LD);
#pragma unroll
      for (int nt = 0; nt < 2; ++nt) {
        v16bf b = load_b_bf16(vp + k0 * VP_LD + nt * 16, VP_LD);
        acc[nt] = wmma_bf16(a, b, acc[nt]);
      }
    }
#pragma unroll
    for (int nt = 0; nt < 2; ++nt) {
#pragma unroll
      for (int r = 0; r < 8; ++r) {
        int n = g * WS_ + s * 16 + r + 8 * half;
        int c = h * D_ + nt * 16 + n_local;
        outw[((long long)bb * N_ + n) * INNER_ + c] = (__bf16)acc[nt][r];
      }
    }
  }
}

// ---------------- Kernel 3: output projection + bias ----------------------
// [65536x256]bf16 @ Wout^T[256x256]; 16x64 strip per wave.
__global__ void __launch_bounds__(256)
out_proj_kernel(const unsigned short* __restrict__ araw,
                const float* __restrict__ Wout, const float* __restrict__ bout,
                float* __restrict__ out) {
  const __bf16* A = reinterpret_cast<const __bf16*>(araw);
  int wave = threadIdx.x >> 5;
  int lane = threadIdx.x & 31;
  int t = blockIdx.x * 8 + wave;          // 4096*4 = 16384 strip jobs
  int strip  = t & 3;                     // 4 col strips of 64
  int tile_m = t >> 2;

  v8f acc[4] = {};
#pragma unroll
  for (int k0 = 0; k0 < INNER_; k0 += 32) {
    v16bf a = load_a_bf16(A + (long long)tile_m * 16 * INNER_ + k0, INNER_);
#pragma unroll
    for (int j = 0; j < 4; ++j) {
      v16bf b = load_bT_f32(Wout + (long long)(strip * 64 + j * 16) * INNER_ + k0, INNER_);
      acc[j] = wmma_bf16(a, b, acc[j]);
    }
  }

  int half = lane >> 4, n_local = lane & 15;
#pragma unroll
  for (int j = 0; j < 4; ++j) {
    int C = strip * 64 + j * 16 + n_local;
    float bias = bout[C];
#pragma unroll
    for (int r = 0; r < 8; ++r) {
      int R = tile_m * 16 + r + 8 * half;
      out[(long long)R * DIM_ + C] = acc[j][r] + bias;
    }
  }
}

// ---------------- host launcher -------------------------------------------
extern "C" void kernel_launch(void* const* d_in, const int* in_sizes, int n_in,
                              void* d_out, int out_size, void* d_ws, size_t ws_size,
                              hipStream_t stream) {
  const float* x    = (const float*)d_in[0];
  const float* Wqkv = (const float*)d_in[1];
  const float* E    = (const float*)d_in[2];
  const float* F    = (const float*)d_in[3];
  const float* Wout = (const float*)d_in[4];
  const float* bout = (const float*)d_in[5];
  float* out        = (float*)d_out;

  unsigned short* ws = (unsigned short*)d_ws;
  unsigned short* qws = ws;
  unsigned short* kws = ws + QSZ_;
  unsigned short* vws = ws + 2 * QSZ_;
  unsigned short* aws = ws + 3 * QSZ_;

  // 4096 row tiles * 12 strips, 8 waves per block
  qkv_proj_kernel<<<49152 / 8, 256, 0, stream>>>(x, Wqkv, qws, kws, vws);
  // one block per (b,h,g)
  attn_kernel<<<B_ * H_ * G_, 256, 0, stream>>>(qws, kws, vws, E, F, aws);
  // 4096 row tiles * 4 strips, 8 waves per block
  out_proj_kernel<<<16384 / 8, 256, 0, stream>>>(aws, Wout, bout, out);
}
